// HeteroGNNEncoder_89464168776241
// MI455X (gfx1250) — matrix operand order repaired and checked by reading, not compiled, not run on hardware
//
#include <hip/hip_runtime.h>

typedef __attribute__((ext_vector_type(16))) __bf16 v16bf;
typedef __attribute__((ext_vector_type(8)))  __bf16 v8bf;
typedef __attribute__((ext_vector_type(8)))  float  v8f;

#define D128 128

// ---------- helpers ----------

__device__ __forceinline__ float atomAdd(float* p, float v) {
  return __hip_atomic_fetch_add(p, v, __ATOMIC_RELAXED, __HIP_MEMORY_SCOPE_AGENT);
}

// f32 -> bf16, round-to-nearest-even (pure bit ops: no reliance on __bf16 arith)
__device__ __forceinline__ __bf16 f2bf(float f) {
  unsigned u = __builtin_bit_cast(unsigned, f);
  u += 0x7FFFu + ((u >> 16) & 1u);
  unsigned short s = (unsigned short)(u >> 16);
  return __builtin_bit_cast(__bf16, s);
}
__device__ __forceinline__ float bf2f(__bf16 b) {
  unsigned short s = __builtin_bit_cast(unsigned short, b);
  unsigned u = ((unsigned)s) << 16;
  return __builtin_bit_cast(float, u);
}

template <typename T> __device__ __forceinline__ float loadf(const T* p);
template <> __device__ __forceinline__ float loadf<float>(const float* p)   { return *p; }
template <> __device__ __forceinline__ float loadf<__bf16>(const __bf16* p) { return bf2f(*p); }

// Build a 16-element bf16 fragment from two 16-byte chunks (2x b128 loads)
__device__ __forceinline__ v16bf frag16(const __bf16* p0, const __bf16* p1) {
  v8bf lo = *(const v8bf*)p0;
  v8bf hi = *(const v8bf*)p1;
  return __builtin_shufflevector(lo, hi, 0,1,2,3,4,5,6,7,8,9,10,11,12,13,14,15);
}

// ---------- small utility kernels ----------

__global__ void zero_f32(float* __restrict__ p, int n) {
  int i = blockIdx.x * blockDim.x + threadIdx.x;
  if (i < n) p[i] = 0.0f;
}

__global__ void count_idx(const int* __restrict__ idx, int n, float* __restrict__ cnt) {
  int i = blockIdx.x * blockDim.x + threadIdx.x;
  int stride = gridDim.x * blockDim.x;
  for (; i < n; i += stride) atomAdd(&cnt[idx[i]], 1.0f);
}

// W [128x128] f32 row-major (k-major) -> Wt [n][k] bf16 (transposed)
__global__ void conv_weight_t(const float* __restrict__ w, __bf16* __restrict__ wt) {
  int i = blockIdx.x * blockDim.x + threadIdx.x;  // 0..16383
  int k = i >> 7, n = i & 127;
  wt[n * D128 + k] = f2bf(w[i]);
}

// per-edge: agg[dst][t] += feat[src][t]  (atomic f32, L2-resident)
template <typename T>
__global__ void scatter_edges(const T* __restrict__ feat, const int* __restrict__ esrc,
                              const int* __restrict__ edst, float* __restrict__ agg, int E) {
  int e = blockIdx.x * 2 + (threadIdx.x >> 7);
  if (e >= E) return;
  int t = threadIdx.x & 127;
  int s = esrc[e], d = edst[e];
  float v = loadf<T>(feat + (size_t)s * D128 + t);
  atomAdd(&agg[(size_t)d * D128 + t], v);
}

// ---------- fused SAGE layer: relu(mean_agg @ Wl + bl + x @ Wr) ----------
// block = 256 threads = 8 wave32; block handles 16 rows; wave w -> N-tile [w*16, w*16+16)
template <typename TDense, bool POOL>
__global__ __launch_bounds__(256) void sage_gemm(
    const float* __restrict__ agg, const float* __restrict__ cnt,
    const TDense* __restrict__ xdense,
    const __bf16* __restrict__ WlT, const __bf16* __restrict__ WrT,  // [n][k] bf16
    const float* __restrict__ bias,
    __bf16* __restrict__ hout,                       // !POOL: [N,128] bf16
    float* __restrict__ pool, const int* __restrict__ batch,  // POOL: [G,128] f32
    int N)
{
  __shared__ __bf16 sA[16][D128];  // mean-aggregated neighbors (bf16)
  __shared__ __bf16 sX[16][D128];  // dense/self features (bf16)

  const int tid = threadIdx.x;
  const int m0  = blockIdx.x * 16;

  // ---- stage: each thread converts 8 contiguous elements of one row ----
  {
    int r   = tid >> 4;          // 0..15
    int c0  = (tid & 15) * 8;    // 0..120
    int row = m0 + r; if (row >= N) row = N - 1;  // clamp (no divergence)
    float inv = 1.0f / fmaxf(cnt[row], 1.0f);
    const float*  ap = agg    + (size_t)row * D128 + c0;
    const TDense* xp = xdense + (size_t)row * D128 + c0;
#pragma unroll
    for (int j = 0; j < 8; ++j) {
      sA[r][c0 + j] = f2bf(ap[j] * inv);
      sX[r][c0 + j] = f2bf(loadf<TDense>(xp + j));
    }
  }
  __syncthreads();

  const int wave = tid >> 5, lane = tid & 31;
  const int half = lane >> 4, lr = lane & 15;
  const int n0 = wave * 16;

  v8f acc = {};
#pragma unroll
  for (int k0 = 0; k0 < D128; k0 += 32) {
    // A fragment (16x32): lane half selects K sub-range per ISA layout
    v16bf aA = frag16(&sA[lr][k0 + half * 8], &sA[lr][k0 + 16 + half * 8]);
    v16bf aX = frag16(&sX[lr][k0 + half * 8], &sX[lr][k0 + 16 + half * 8]);
    // B fragment (32x16): column n0+lr, K sub-ranges matching A (half*8, 16+half*8) from transposed weights
    const __bf16* bl = WlT + (size_t)(n0 + lr) * D128 + k0 + half * 8;
    const __bf16* br = WrT + (size_t)(n0 + lr) * D128 + k0 + half * 8;
    v16bf bL = frag16(bl, bl + 16);
    v16bf bR = frag16(br, br + 16);
    acc = __builtin_amdgcn_wmma_f32_16x16x32_bf16(false, aA, false, bL, (short)0, acc, false, false);
    acc = __builtin_amdgcn_wmma_f32_16x16x32_bf16(false, aX, false, bR, (short)0, acc, false, false);
  }

  // ---- epilogue: C layout lane->(row = m0 + half*8 + r, col = n0 + lr) ----
  const int col = n0 + lr;
  const float b = bias[col];
#pragma unroll
  for (int r = 0; r < 8; ++r) {
    int row = m0 + half * 8 + r;
    if (row < N) {
      float v = fmaxf(acc[r] + b, 0.0f);
      if (POOL) atomAdd(&pool[(size_t)batch[row] * D128 + col], v);
      else      hout[(size_t)row * D128 + col] = f2bf(v);
    }
  }
}

// ---------- output head: emb = poolA/cA + poolB/cB ; out = emb @ Wout + bout ----------
__global__ __launch_bounds__(128) void final_out(
    const float* __restrict__ pa, const float* __restrict__ ca,
    const float* __restrict__ pb, const float* __restrict__ cb,
    const float* __restrict__ Wout, const float* __restrict__ bout,
    float* __restrict__ out)
{
  __shared__ float emb[D128];
  int g = blockIdx.x, n = threadIdx.x;
  emb[n] = pa[g * D128 + n] / fmaxf(ca[g], 1.0f) + pb[g * D128 + n] / fmaxf(cb[g], 1.0f);
  __syncthreads();
  float acc = bout[n];
#pragma unroll 4
  for (int k = 0; k < D128; ++k) acc += emb[k] * Wout[k * D128 + n];
  out[g * D128 + n] = acc;
}

// ---------- driver ----------

extern "C" void kernel_launch(void* const* d_in, const int* in_sizes, int n_in,
                              void* d_out, int out_size, void* d_ws, size_t ws_size,
                              hipStream_t stream) {
  (void)n_in; (void)ws_size;
  const float* x_a   = (const float*)d_in[0];
  const float* x_b   = (const float*)d_in[1];
  const int*   e_ab  = (const int*)d_in[2];   // [0,E)=src(A), [E,2E)=dst(B)
  const int*   e_ba  = (const int*)d_in[3];   // [0,E)=src(B), [E,2E)=dst(A)
  const int*   bat_a = (const int*)d_in[4];
  const int*   bat_b = (const int*)d_in[5];
  const float* W1abL = (const float*)d_in[7];
  const float* b1ab  = (const float*)d_in[8];
  const float* W1abR = (const float*)d_in[9];
  const float* W1baL = (const float*)d_in[10];
  const float* b1ba  = (const float*)d_in[11];
  const float* W1baR = (const float*)d_in[12];
  const float* W2abL = (const float*)d_in[13];
  const float* b2ab  = (const float*)d_in[14];
  const float* W2abR = (const float*)d_in[15];
  const float* W2baL = (const float*)d_in[16];
  const float* b2ba  = (const float*)d_in[17];
  const float* W2baR = (const float*)d_in[18];
  const float* Wout  = (const float*)d_in[19];
  const float* bout  = (const float*)d_in[20];

  const int NA = in_sizes[0] / D128;
  const int NB = in_sizes[1] / D128;
  const int E  = in_sizes[2] / 2;
  const int G  = out_size / D128;
  const int NMAX = NA > NB ? NA : NB;

  size_t off = 0;
  char* base = (char*)d_ws;
  auto carve = [&](size_t bytes) -> char* {
    char* p = base + off;
    off += (bytes + 255) & ~(size_t)255;
    return p;
  };
  float*  agg   = (float*)carve((size_t)NMAX * D128 * 4);
  float*  cntAB = (float*)carve((size_t)NB * 4);          // in-degree of B (edge_ab)
  float*  cntBA = (float*)carve((size_t)NA * 4);          // in-degree of A (edge_ba)
  __bf16* h_a   = (__bf16*)carve((size_t)NA * D128 * 2);
  __bf16* h_b   = (__bf16*)carve((size_t)NB * D128 * 2);
  __bf16* Wt    = (__bf16*)carve((size_t)8 * D128 * D128 * 2);
  float*  poolA = (float*)carve((size_t)G * D128 * 4);
  float*  poolB = (float*)carve((size_t)G * D128 * 4);
  float*  cgA   = (float*)carve((size_t)G * 4);
  float*  cgB   = (float*)carve((size_t)G * 4);

  auto Z = [&](float* p, size_t n) {
    zero_f32<<<(unsigned)((n + 255) / 256), 256, 0, stream>>>(p, (int)n);
  };

  // counts (edge in-degrees and nodes-per-graph) -- depend only on topology
  Z(cntAB, NB); Z(cntBA, NA);
  Z(poolA, (size_t)G * D128); Z(poolB, (size_t)G * D128);
  Z(cgA, G); Z(cgB, G);
  count_idx<<<512, 256, 0, stream>>>(e_ab + E, E, cntAB);
  count_idx<<<512, 256, 0, stream>>>(e_ba + E, E, cntBA);
  count_idx<<<256, 256, 0, stream>>>(bat_a, NA, cgA);
  count_idx<<<256, 256, 0, stream>>>(bat_b, NB, cgB);

  // pre-convert the 8 layer weights to transposed bf16
  const float* wsrc[8] = {W1abL, W1abR, W1baL, W1baR, W2abL, W2abR, W2baL, W2baR};
  for (int i = 0; i < 8; ++i)
    conv_weight_t<<<64, 256, 0, stream>>>(wsrc[i], Wt + (size_t)i * D128 * D128);

  const unsigned eb = (unsigned)((E + 1) / 2);

  // layer 1: h_b = relu(mean_{A->B}(x_a) @ Wl + bl + x_b @ Wr)
  Z(agg, (size_t)NB * D128);
  scatter_edges<float><<<eb, 256, 0, stream>>>(x_a, e_ab, e_ab + E, agg, E);
  sage_gemm<float, false><<<(NB + 15) / 16, 256, 0, stream>>>(
      agg, cntAB, x_b, Wt + 0 * 16384, Wt + 1 * 16384, b1ab, h_b, nullptr, nullptr, NB);

  // layer 1: h_a
  Z(agg, (size_t)NA * D128);
  scatter_edges<float><<<eb, 256, 0, stream>>>(x_b, e_ba, e_ba + E, agg, E);
  sage_gemm<float, false><<<(NA + 15) / 16, 256, 0, stream>>>(
      agg, cntBA, x_a, Wt + 2 * 16384, Wt + 3 * 16384, b1ba, h_a, nullptr, nullptr, NA);

  // layer 2: g_b, pooled directly into poolB (never materialized)
  Z(agg, (size_t)NB * D128);
  scatter_edges<__bf16><<<eb, 256, 0, stream>>>(h_a, e_ab, e_ab + E, agg, E);
  sage_gemm<__bf16, true><<<(NB + 15) / 16, 256, 0, stream>>>(
      agg, cntAB, h_b, Wt + 4 * 16384, Wt + 5 * 16384, b2ab, nullptr, poolB, bat_b, NB);

  // layer 2: g_a -> poolA
  Z(agg, (size_t)NA * D128);
  scatter_edges<__bf16><<<eb, 256, 0, stream>>>(h_b, e_ba, e_ba + E, agg, E);
  sage_gemm<__bf16, true><<<(NA + 15) / 16, 256, 0, stream>>>(
      agg, cntBA, h_a, Wt + 6 * 16384, Wt + 7 * 16384, b2ba, nullptr, poolA, bat_a, NA);

  // output head
  final_out<<<G, 128, 0, stream>>>(poolA, cgA, poolB, cgB, Wout, bout, (float*)d_out);
}
